// GCNN_19035295056161
// MI455X (gfx1250) — compile-verified, matrix operand order checked by reading, not run on hardware
//
#include <hip/hip_runtime.h>
#include <hip/hip_bf16.h>

typedef __attribute__((ext_vector_type(16))) __bf16 v16bf;
typedef __attribute__((ext_vector_type(8)))  float  v8f;

#define FDIM 64
#define NEG_SLOPE 0.01f

// ---------------------------------------------------------------------------
// Degree / norm precompute
// ---------------------------------------------------------------------------
__global__ void k_deg_init(float* __restrict__ deg, int N) {
    int i = blockIdx.x * blockDim.x + threadIdx.x;
    if (i < N) deg[i] = 1.0f;   // self-loop
}

__global__ void k_deg_accum(const int* __restrict__ dst, float* __restrict__ deg, int E) {
    int e = blockIdx.x * blockDim.x + threadIdx.x;
    if (e < E) atomicAdd(&deg[dst[e]], 1.0f);
}

__global__ void k_dinv(const float* __restrict__ deg, float* __restrict__ dinv, int N) {
    int i = blockIdx.x * blockDim.x + threadIdx.x;
    if (i < N) {
        float d = deg[i];
        dinv[i] = (d > 0.0f) ? rsqrtf(d) : 0.0f;
    }
}

__global__ void k_norm(const int* __restrict__ src, const int* __restrict__ dst,
                       const float* __restrict__ dinv, float* __restrict__ nrm, int E) {
    int e = blockIdx.x * blockDim.x + threadIdx.x;
    if (e < E) nrm[e] = dinv[src[e]] * dinv[dst[e]];
}

// ---------------------------------------------------------------------------
// WMMA GEMM + fused epilogue:
//   H[N,64]   = f( X[N,K] (+bias, LeakyReLU) ) @ W[K,64]
//   AGG[N,64] = H * dinv^2         (self-loop init for the aggregation)
//
// Block = 256 threads = 8 waves. W is staged once per block into LDS with
// gfx1250 async global->LDS copies (ASYNCcnt). Each wave owns a 16-row tile
// and computes all four 16-col tiles from one A fragment (4 accumulators),
// issuing 4 v_wmma_f32_16x16x32_bf16 per K-step. Fragment layouts per CDNA5
// ISA 7.12.2 (wave32).
// ---------------------------------------------------------------------------
template<int K, bool FUSE_BIAS_LRELU>
__global__ __launch_bounds__(256) void k_gemm_wmma(
    const float* __restrict__ X, const float* __restrict__ W,
    const float* __restrict__ bias_in, const float* __restrict__ dinv,
    float* __restrict__ H, float* __restrict__ AGG, int N)
{
    __shared__ float Wl[K * FDIM];

    // ---- stage W[K,64] (f32) into LDS via async copy (16B per lane per op)
    for (int i = threadIdx.x * 4; i < K * FDIM; i += 256 * 4) {
        unsigned     lds_off = (unsigned)(uintptr_t)(&Wl[i]);  // low 32 bits = LDS byte addr
        const float* gsrc    = W + i;
        asm volatile("global_load_async_to_lds_b128 %0, %1, off"
                     :: "v"(lds_off), "v"(gsrc) : "memory");
    }
    asm volatile("s_wait_asynccnt 0x0" ::: "memory");
    __syncthreads();

    const int lane = threadIdx.x & 31;
    const int wave = threadIdx.x >> 5;
    const int row0 = (blockIdx.x * 8 + wave) * 16;
    if (row0 >= N) return;

    const int nl    = lane & 15;
    int m = row0 + nl;
    if (m >= N) m = N - 1;                       // clamp loads; store is guarded
    const int kbase = (lane < 16) ? 0 : 8;       // K sub-offset for this lane half

    v8f c0 = {}, c1 = {}, c2 = {}, c3 = {};

    #pragma unroll
    for (int k0 = 0; k0 < K; k0 += 32) {
        // ---- A fragment: row m, K = k0+kbase+{0..7} and k0+16+kbase+{0..7}
        const float* xr = X + (size_t)m * K + k0 + kbase;
        float4 x0 = *(const float4*)(xr + 0);
        float4 x1 = *(const float4*)(xr + 4);
        float4 x2 = *(const float4*)(xr + 16);
        float4 x3 = *(const float4*)(xr + 20);
        float xa[16] = { x0.x, x0.y, x0.z, x0.w,  x1.x, x1.y, x1.z, x1.w,
                         x2.x, x2.y, x2.z, x2.w,  x3.x, x3.y, x3.z, x3.w };

        v16bf a;
        #pragma unroll
        for (int j = 0; j < 16; ++j) {
            int kk = k0 + kbase + ((j < 8) ? j : (j + 8));
            float v = xa[j];
            if (FUSE_BIAS_LRELU) {
                v += bias_in[kk];
                v = (v >= 0.0f) ? v : NEG_SLOPE * v;   // LeakyReLU
            }
            a[j] = (__bf16)v;
        }

        // ---- four column tiles share this A fragment
        #pragma unroll
        for (int t = 0; t < 4; ++t) {
            v16bf b;
            #pragma unroll
            for (int j = 0; j < 16; ++j) {
                int kk = k0 + kbase + ((j < 8) ? j : (j + 8));
                b[j] = (__bf16)Wl[kk * FDIM + t * 16 + nl];
            }
            v8f& c = (t == 0) ? c0 : (t == 1) ? c1 : (t == 2) ? c2 : c3;
            c = __builtin_amdgcn_wmma_f32_16x16x32_bf16(
                    false, a, false, b, (short)0, c, false, false);
        }
    }

    // ---- fused store: H and AGG = H * dinv^2
    const int moff = (lane >> 4) << 3;           // +8 rows for upper lane half
    #pragma unroll
    for (int r = 0; r < 8; ++r) {
        int mm = row0 + r + moff;
        if (mm < N) {
            float di = dinv[mm];
            float w  = di * di;
            float vr[4] = { c0[r], c1[r], c2[r], c3[r] };
            #pragma unroll
            for (int t = 0; t < 4; ++t) {
                size_t o = (size_t)mm * FDIM + t * 16 + nl;
                H[o]   = vr[t];
                AGG[o] = vr[t] * w;
            }
        }
    }
}

// ---------------------------------------------------------------------------
// Edge scatter: agg[dst] += norm * h[src].  16 threads/row, float4 each.
// ---------------------------------------------------------------------------
__global__ void k_scatter(const int* __restrict__ src, const int* __restrict__ dst,
                          const float* __restrict__ nrm, const float* __restrict__ h,
                          float* __restrict__ agg, int E) {
    int g = blockIdx.x * blockDim.x + threadIdx.x;
    if (g >= E * 16) return;
    int e = g >> 4;
    int f = (g & 15) << 2;
    int s = src[e];
    int d = dst[e];
    float w = nrm[e];
    float4 v = *(const float4*)(h + (size_t)s * FDIM + f);
    float* o = agg + (size_t)d * FDIM + f;
    atomicAdd(o + 0, v.x * w);
    atomicAdd(o + 1, v.y * w);
    atomicAdd(o + 2, v.z * w);
    atomicAdd(o + 3, v.w * w);
}

// ---------------------------------------------------------------------------
// FC: out[N,4] = (agg2 + b2) @ Wfc + bfc.  One wave per node, shfl reduction.
// ---------------------------------------------------------------------------
__global__ __launch_bounds__(256) void k_fc(const float* __restrict__ A,
                                            const float* __restrict__ b2,
                                            const float* __restrict__ Wfc,
                                            const float* __restrict__ bfc,
                                            float* __restrict__ out, int N) {
    int node = blockIdx.x * (blockDim.x >> 5) + (threadIdx.x >> 5);
    int lane = threadIdx.x & 31;
    if (node >= N) return;

    float h0 = A[(size_t)node * FDIM + lane]      + b2[lane];
    float h1 = A[(size_t)node * FDIM + 32 + lane] + b2[32 + lane];

    float acc[4];
    #pragma unroll
    for (int cc = 0; cc < 4; ++cc)
        acc[cc] = h0 * Wfc[lane * 4 + cc] + h1 * Wfc[(32 + lane) * 4 + cc];

    #pragma unroll
    for (int off = 16; off > 0; off >>= 1) {
        #pragma unroll
        for (int cc = 0; cc < 4; ++cc)
            acc[cc] += __shfl_xor(acc[cc], off, 32);
    }
    if (lane == 0) {
        #pragma unroll
        for (int cc = 0; cc < 4; ++cc)
            out[(size_t)node * 4 + cc] = acc[cc] + bfc[cc];
    }
}

// ---------------------------------------------------------------------------
extern "C" void kernel_launch(void* const* d_in, const int* in_sizes, int n_in,
                              void* d_out, int out_size, void* d_ws, size_t ws_size,
                              hipStream_t stream) {
    const float* x    = (const float*)d_in[0];
    const int*   ei   = (const int*)  d_in[1];
    const float* W1   = (const float*)d_in[2];
    const float* b1   = (const float*)d_in[3];
    const float* W2   = (const float*)d_in[4];
    const float* b2   = (const float*)d_in[5];
    const float* Wfc  = (const float*)d_in[6];
    const float* bfc  = (const float*)d_in[7];
    float*       out  = (float*)d_out;

    const int N = in_sizes[0] / 128;
    const int E = in_sizes[1] / 2;
    const int* src = ei;
    const int* dst = ei + E;

    // workspace carve-out
    float* ws   = (float*)d_ws;
    float* deg  = ws;                         // N
    float* dinv = deg  + N;                   // N
    float* nrm  = dinv + N;                   // E
    float* h1   = nrm  + E;                   // N*64
    float* agg1 = h1   + (size_t)N * FDIM;    // N*64
    float* h2   = agg1 + (size_t)N * FDIM;    // N*64
    float* agg2 = h2   + (size_t)N * FDIM;    // N*64

    const int T = 256;
    dim3 blk(T);

    // degree / normalization coefficients
    k_deg_init <<<dim3((N + T - 1) / T), blk, 0, stream>>>(deg, N);
    k_deg_accum<<<dim3((E + T - 1) / T), blk, 0, stream>>>(dst, deg, E);
    k_dinv     <<<dim3((N + T - 1) / T), blk, 0, stream>>>(deg, dinv, N);
    k_norm     <<<dim3((E + T - 1) / T), blk, 0, stream>>>(src, dst, dinv, nrm, E);

    // layer 1: h1 = x @ W1 ; agg1 = h1*dinv^2 (fused) ; += edge messages
    k_gemm_wmma<128, false><<<dim3((N + 127) / 128), dim3(256), 0, stream>>>(
        x, W1, nullptr, dinv, h1, agg1, N);
    k_scatter<<<dim3((E * 16 + T - 1) / T), blk, 0, stream>>>(src, dst, nrm, h1, agg1, E);

    // layer 2: h2 = LeakyReLU(agg1 + b1) @ W2 ; agg2 = h2*dinv^2 ; += messages
    k_gemm_wmma<64, true><<<dim3((N + 127) / 128), dim3(256), 0, stream>>>(
        agg1, W2, b1, dinv, h2, agg2, N);
    k_scatter<<<dim3((E * 16 + T - 1) / T), blk, 0, stream>>>(src, dst, nrm, h2, agg2, E);

    // final FC: out = (agg2 + b2) @ Wfc + bfc
    k_fc<<<dim3((N + 7) / 8), dim3(256), 0, stream>>>(agg2, b2, Wfc, bfc, out, N);
}
